// CompressAttn_42812234007031
// MI455X (gfx1250) — matrix operand order repaired and checked by reading, not compiled, not run on hardware
//
#include <hip/hip_runtime.h>

// ---------------- problem constants (fixed by setup_inputs) ----------------
#define B_   2
#define N_   4096
#define QH_  32
#define KH_  2
#define G_   16      // QH/KH
#define D_   128
#define KS_  32      // kernel_size
#define ST_  16      // stride
#define M_   255     // (N-KS)/ST + 1
#define MP_  256     // padded M (row 255 always masked: 255*16+31 > 4095)

typedef _Float16 v16h __attribute__((ext_vector_type(16)));
typedef _Float16 v8h  __attribute__((ext_vector_type(8)));
typedef float    v8f  __attribute__((ext_vector_type(8)));
typedef float    v4f  __attribute__((ext_vector_type(4)));

// ---------------------------------------------------------------------------
// Kernel 1: compress k,v -> ck [b][h][m][d] (f16)  and  cv^T [b][h][d][m] (f16)
// ck[b,h,m,d] = sum_k w[k]*(x[b, m*ST+k, h, d] + pe[k,d]) / max(sum w, 1e-6)
// ---------------------------------------------------------------------------
__global__ __launch_bounds__(256)
void ca_compress(const float* __restrict__ kin, const float* __restrict__ vin,
                 const float* __restrict__ wk,  const float* __restrict__ pek,
                 const float* __restrict__ wv,  const float* __restrict__ pev,
                 _Float16* __restrict__ ckh, _Float16* __restrict__ cvTh)
{
    int tid = blockIdx.x * blockDim.x + threadIdx.x;   // B*KH*MP*D threads
    int d = tid & (D_ - 1);
    int m = (tid >> 7) & (MP_ - 1);
    int h = (tid >> 15) & (KH_ - 1);
    int b = tid >> 16;

    float swk = 0.f, swv = 0.f;
    #pragma unroll
    for (int kk = 0; kk < KS_; ++kk) { swk += wk[kk]; swv += wv[kk]; }

    float ak = 0.f, av = 0.f;
    if (m < M_) {
        const size_t base = (((size_t)b * N_ + (size_t)m * ST_) * KH_ + h) * D_ + d;
        const size_t step = (size_t)KH_ * D_;
        #pragma unroll 4
        for (int kk = 0; kk < KS_; ++kk) {
            ak += wk[kk] * (kin[base + (size_t)kk * step] + pek[kk * D_ + d]);
            av += wv[kk] * (vin[base + (size_t)kk * step] + pev[kk * D_ + d]);
        }
        ak /= fmaxf(swk, 1e-6f);
        av /= fmaxf(swv, 1e-6f);
    }
    const int bh = b * KH_ + h;
    ckh [((size_t)bh * MP_ + m) * D_  + d] = (_Float16)ak;
    cvTh[((size_t)bh * D_  + d) * MP_ + m] = (_Float16)av;
}

// ---------------------------------------------------------------------------
// Kernel 2: flash attention with WMMA f16, one wave per 16-query tile.
//   S^T(16m x 16n) = A(ck rows) x B(Q^T),  O^T(16d x 16n) += A(cv^T rows) x B(P^T)
// Fragment layouts used here (K order contiguous in A and B; ISA table order differs):
//   A 16x32 f16: lane<16 row=lp, K halves {o1..o1+7}={half*16..}, {o2..o2+7}={half*16+8..}
//   B 32x16 f16: lane<16 col=lp holds K=0..15; lane>=16 col=lp holds K=16..31
//   C/D f32:     VGPR i = row (i + 8*half), lane -> col lp
// ---------------------------------------------------------------------------
__global__ __launch_bounds__(256)
void ca_attn(const float* __restrict__ q,
             const _Float16* __restrict__ ckh,
             const _Float16* __restrict__ cvTh,
             float* __restrict__ out)
{
    __shared__ __align__(16) float lds[8][16 * 128];   // 64 KB, rotate-swizzled

    const int wave = threadIdx.x >> 5;
    const int lane = threadIdx.x & 31;
    const int half = lane >> 4;
    const int lp   = lane & 15;

    const int wid = blockIdx.x * 8 + wave;   // 16384 waves total
    const int t = wid & 255;                 // n-tile index (n0 = 16*t)
    const int g = (wid >> 8) & 15;
    const int h = (wid >> 12) & 1;
    const int b = wid >> 13;

    const int n0 = t * 16;
    const int n  = n0 + lp;                  // this lane's query column
    const int qh = h * G_ + g;
    const int bh = b * KH_ + h;
    const float sm_scale = 0.08838834764831845f;   // D^-0.5

    // ---- Q^T B-fragments: lane = column n, K(d) = dc*32 + half*16 + j ----
    const float* qrow = q + (((size_t)b * N_ + n) * QH_ + qh) * D_;
    v16h qf[4];
    #pragma unroll
    for (int dc = 0; dc < 4; ++dc) {
        const float* p = qrow + dc * 32 + half * 16;
        #pragma unroll
        for (int j = 0; j < 16; ++j) qf[dc][j] = (_Float16)p[j];
    }

    v8f acc[8];
    #pragma unroll
    for (int dg = 0; dg < 8; ++dg) acc[dg] = (v8f){};

    float mrun = -1e30f, lrun = 0.f;

    // tile t attends to m in [0, t-1]  (clamped to M-1 by padding+mask)
    const int nm = t;                       // number of candidate m values
    const int nchunks = (nm + 31) >> 5;
    const _Float16* kbase = ckh  + (size_t)bh * MP_ * D_;
    const _Float16* vbase = cvTh + (size_t)bh * D_ * MP_;

    for (int c = 0; c < nchunks; ++c) {
        const int m0 = c * 32;

        // ---- S^T = ck x Q^T : two 16x16 tiles (m0..15, m0+16..31) ----
        v8f s0 = (v8f){}, s1 = (v8f){};
        const _Float16* kr0 = kbase + (size_t)(m0 + lp) * D_;
        const _Float16* kr1 = kr0 + (size_t)16 * D_;
        #pragma unroll
        for (int dc = 0; dc < 4; ++dc) {
            const int o1 = dc * 32 + half * 16;
            const int o2 = o1 + 8;
            v16h a0, a1;
            {
                v8h x = *(const v8h*)(kr0 + o1);
                v8h y = *(const v8h*)(kr0 + o2);
                #pragma unroll
                for (int j = 0; j < 8; ++j) { a0[j] = x[j]; a0[8 + j] = y[j]; }
                x = *(const v8h*)(kr1 + o1);
                y = *(const v8h*)(kr1 + o2);
                #pragma unroll
                for (int j = 0; j < 8; ++j) { a1[j] = x[j]; a1[8 + j] = y[j]; }
            }
            s0 = __builtin_amdgcn_wmma_f32_16x16x32_f16(false, a0, false, qf[dc],
                                                        (short)0, s0, false, false);
            s1 = __builtin_amdgcn_wmma_f32_16x16x32_f16(false, a1, false, qf[dc],
                                                        (short)0, s1, false, false);
        }

        // ---- scale + mask + online softmax (stats per column n) ----
        float p0[8], p1[8];
        float mx = -1e30f;
        #pragma unroll
        for (int i = 0; i < 8; ++i) {
            const int mi0 = m0 + 8 * half + i;       // row of s0[i]
            const int mi1 = mi0 + 16;                // row of s1[i]
            float v0 = (n >= mi0 * 16 + 31) ? s0[i] * sm_scale : -1e30f;
            float v1 = (n >= mi1 * 16 + 31) ? s1[i] * sm_scale : -1e30f;
            p0[i] = v0; p1[i] = v1;
            mx = fmaxf(mx, fmaxf(v0, v1));
        }
        mx = fmaxf(mx, __shfl_xor(mx, 16));          // combine lane halves
        const float mnew = fmaxf(mrun, mx);
        const float corr = __expf(mrun - mnew);
        mrun = mnew;

        float ls = 0.f;
        #pragma unroll
        for (int i = 0; i < 8; ++i) {
            const int mi0 = m0 + 8 * half + i;
            const int mi1 = mi0 + 16;
            float e0 = (n >= mi0 * 16 + 31) ? __expf(p0[i] - mnew) : 0.f;
            float e1 = (n >= mi1 * 16 + 31) ? __expf(p1[i] - mnew) : 0.f;
            p0[i] = e0; p1[i] = e1;
            ls += e0 + e1;
        }
        ls += __shfl_xor(ls, 16);
        lrun = lrun * corr + ls;

        #pragma unroll
        for (int dg = 0; dg < 8; ++dg)
            #pragma unroll
            for (int i = 0; i < 8; ++i) acc[dg][i] *= corr;

        // ---- build P^T B-fragment: col n, K(m)=half*16+j; swap halves via xor16 ----
        v16h pf;
        #pragma unroll
        for (int i = 0; i < 8; ++i) {
            float x0 = __shfl_xor(p0[i], 16);
            float x1 = __shfl_xor(p1[i], 16);
            pf[i]     = (_Float16)(half ? x1    : p0[i]);
            pf[8 + i] = (_Float16)(half ? p1[i] : x0);
        }

        // ---- O^T += cv^T x P^T : one WMMA per 16-d group ----
        #pragma unroll
        for (int dg = 0; dg < 8; ++dg) {
            const _Float16* vr = vbase + (size_t)(dg * 16 + lp) * MP_ + m0;
            v16h a;
            v8h x = *(const v8h*)(vr + half * 16);
            v8h y = *(const v8h*)(vr + half * 16 + 8);
            #pragma unroll
            for (int j = 0; j < 8; ++j) { a[j] = x[j]; a[8 + j] = y[j]; }
            acc[dg] = __builtin_amdgcn_wmma_f32_16x16x32_f16(false, a, false, pf,
                                                             (short)0, acc[dg],
                                                             false, false);
        }
    }

    // ---- normalize, transpose via swizzled LDS, coalesced store ----
    const float inv = (lrun > 0.f) ? (1.f / lrun) : 0.f;
    #pragma unroll
    for (int dg = 0; dg < 8; ++dg)
        #pragma unroll
        for (int i = 0; i < 8; ++i) {
            const int d = dg * 16 + 8 * half + i;
            lds[wave][lp * 128 + ((d + lp * 8) & 127)] = acc[dg][i] * inv;
        }
    __syncthreads();

    #pragma unroll 4
    for (int r = 0; r < 16; ++r) {
        const v4f val = *(const v4f*)&lds[wave][r * 128 + ((lane * 4 + r * 8) & 127)];
        float* dst = out + (((size_t)b * N_ + (n0 + r)) * QH_ + qh) * D_ + lane * 4;
        *(v4f*)dst = val;
    }
}

// ---------------------------------------------------------------------------
extern "C" void kernel_launch(void* const* d_in, const int* in_sizes, int n_in,
                              void* d_out, int out_size, void* d_ws, size_t ws_size,
                              hipStream_t stream)
{
    (void)in_sizes; (void)n_in; (void)out_size; (void)ws_size;
    const float* q   = (const float*)d_in[0];
    const float* k   = (const float*)d_in[1];
    const float* v   = (const float*)d_in[2];
    const float* wk  = (const float*)d_in[3];
    const float* pek = (const float*)d_in[4];
    const float* wv  = (const float*)d_in[5];
    const float* pev = (const float*)d_in[6];

    _Float16* ckh  = (_Float16*)d_ws;                       // B*KH*MP*D f16 = 256 KB
    _Float16* cvTh = ckh + (size_t)B_ * KH_ * MP_ * D_;     // +256 KB

    // 1) compress: B*KH*MP*D = 131072 threads
    ca_compress<<<512, 256, 0, stream>>>(k, v, wk, pek, wv, pev, ckh, cvTh);

    // 2) attention: 16384 waves (B*KH*G*(N/16)), 8 waves per block
    ca_attn<<<2048, 256, 0, stream>>>(q, ckh, cvTh, (float*)d_out);
}
